// SAKELayer_48387101556867
// MI455X (gfx1250) — compile-verified, matrix operand order checked by reading, not run on hardware
//
#include <hip/hip_runtime.h>

// ---------------------------------------------------------------------------
// SAKE layer for MI455X (gfx1250, wave32, WMMA), transposed formulation:
// every per-edge GEMM computed as X^T_out = W^T @ X^T_in so the f32 WMMA
// accumulator (D layout) converts lane-locally (v_cvt_pk_f16_f32) into the
// f16 B operand of the next GEMM -- no LDS staging, no barriers in the loop.
// An opaque zero offset (asm) keeps the loop-invariant LDS reads (weights,
// biases) from being LICM-hoisted into ~400 registers: they reload from LDS
// each iteration, keeping VGPR pressure low enough for 2-4 waves/SIMD.
// ---------------------------------------------------------------------------

typedef _Float16 half_t;
typedef __attribute__((ext_vector_type(16))) _Float16 v16h;
typedef __attribute__((ext_vector_type(8)))  float    v8f;

#define TOTN 2048                 // 8 batches * 256 nodes
#define XOFF (TOTN * 64)          // h_new elements; x_new follows in d_out

__device__ __forceinline__ float fast_rcp(float v) { return __builtin_amdgcn_rcpf(v); }
__device__ __forceinline__ float silu_f(float v) {
  return v * fast_rcp(1.0f + __expf(-v));
}
__device__ __forceinline__ float tanh_f(float v) {
  return 1.0f - 2.0f * fast_rcp(__expf(2.0f * v) + 1.0f);
}

__device__ __forceinline__ v8f wmma_f16(v16h a, v16h b, v8f c) {
  return __builtin_amdgcn_wmma_f32_16x16x32_f16(false, a, false, b, (short)0, c,
                                                false, false);
}

// Prepacked W^T A-operand: contiguous 16 halves per lane (2x ds_load_b128).
__device__ __forceinline__ v16h load_W(const half_t* wl, int zo, int mat, int nt,
                                       int ks, int lane) {
  return *(const v16h*)(wl + zo + ((((mat * 4 + nt) * 2 + ks) * 32 + lane) * 16));
}

// ---------------------------------------------------------------------------
// K0: pack four 64x64 matrices into f16 WMMA A-operand lane layout (as W^T).
//   mat0: df_out_w1 (50x64; K=63 row carries df_out_b1, rest of pad = 0)
//   mat1: df_out_w2   mat2: ew_w rows 128..191   mat3: cm_w1
// element: lane = Mlocal + 16*hb, half t -> K = 32ks + (t<8 ? 8hb+t : 16+8hb+t-8)
// ---------------------------------------------------------------------------
__global__ void sake_pack_weights(const float* w1, const float* b1, const float* w2,
                                  const float* ew_w, const float* cm1, half_t* wp) {
  const float* src[4] = {w1, w2, ew_w + 128 * 64, cm1};
  for (int e = threadIdx.x; e < 4096; e += blockDim.x) {
    const int t  = e & 15;
    const int ln = (e >> 4) & 31;
    const int ks = (e >> 9) & 1;
    const int nt = (e >> 10) & 3;
    const int hb = (ln >> 4) & 1;
    const int K  = ks * 32 + ((t >> 3) << 4) + hb * 8 + (t & 7);
    const int N  = nt * 16 + (ln & 15);   // out-feature = A row M
#pragma unroll
    for (int m = 0; m < 4; ++m) {
      float v;
      if (m == 0 && K >= 50) v = (K == 63) ? b1[N] : 0.0f;
      else                   v = src[m][K * 64 + N];
      wp[m * 4096 + e] = (half_t)v;
    }
  }
}

// ---------------------------------------------------------------------------
// K1: per-node precompute of the factored concat-GEMMs (tiny VALU work).
// ---------------------------------------------------------------------------
__global__ __launch_bounds__(64) void sake_node_pre(
    const float* h, const float* df_in_w, const float* df_in_b,
    const float* ew_w, const float* ew_b, const float* sa_w,
    float* pa, float* pb, float* ea, float* eb, float* sai, float* saj) {
  __shared__ float hs[64];
  const int node = blockIdx.x;
  const int t = threadIdx.x;
  hs[t] = h[node * 64 + t];
  __syncthreads();

  float a1 = 0.f, a2 = 0.f;
  if (t < 50) {
#pragma unroll 4
    for (int d = 0; d < 64; ++d) {
      a1 += hs[d] * df_in_w[d * 50 + t];
      a2 += hs[d] * df_in_w[(64 + d) * 50 + t];
    }
    a1 += df_in_b[t];
  }
  pa[node * 64 + t] = (t < 50) ? a1 : 0.f;
  pb[node * 64 + t] = (t < 50) ? a2 : 0.f;

  float a3 = 0.f, a4 = 0.f;
#pragma unroll 4
  for (int d = 0; d < 64; ++d) {
    a3 += hs[d] * ew_w[d * 64 + t];
    a4 += hs[d] * ew_w[(64 + d) * 64 + t];
  }
  ea[node * 64 + t] = a3 + ew_b[t];
  eb[node * 64 + t] = a4;

  if (t == 0) { float s = 0.f; for (int d = 0; d < 64; ++d) s += hs[d] * sa_w[d];      sai[node] = s; }
  if (t == 1) { float s = 0.f; for (int d = 0; d < 64; ++d) s += hs[d] * sa_w[64 + d]; saj[node] = s; }
}

// ---------------------------------------------------------------------------
// K2: edge kernel. One wave32 per query node i (lane&15 = edge in the j-tile,
// lane>>4 = feature-half). 4 waves/block; a single block barrier (weights).
// ---------------------------------------------------------------------------
struct WaveShm {
  float stage[16 * 64];   // coeff tile (edge-major) for the comb reduction
  float att[256];
  float pai[64];
  float eai[64];
  float dxu[16][3];
};

__global__ __launch_bounds__(128) void sake_edges(
    const float* x, const float* df_out_b2,
    const float* cm_b1, const float* cm_w2, const float* cm_b2,
    const float* pa, const float* pb, const float* ea, const float* eb,
    const float* sai, const float* saj, const half_t* wpack_g,
    float* hagg_o, float* comb_o, float* xacc_o) {
  __shared__ __align__(32) WaveShm sw[4];
  __shared__ __align__(32) half_t wpk[4 * 4096];
  __shared__ float b2l[64], cb1l[64], w2l[64];

  const int tid  = threadIdx.x;
  const int wv   = tid >> 5;
  const int lane = tid & 31;
  const int node = blockIdx.x * 4 + wv;
  const int bb   = node >> 8;
  const int i    = node & 255;
  WaveShm& S = sw[wv];

  {  // block-wide: prepacked weights global -> LDS (8192 dwords)
    const unsigned* s = (const unsigned*)wpack_g;
    unsigned* d = (unsigned*)wpk;
    for (int k = tid; k < 8192; k += 128) d[k] = s[k];
  }
  if (tid < 64) { b2l[tid] = df_out_b2[tid]; cb1l[tid] = cm_b1[tid]; w2l[tid] = cm_w2[tid]; }

  for (int t = lane; t < 64; t += 32) { S.pai[t] = pa[node * 64 + t]; S.eai[t] = ea[node * 64 + t]; }
  const float my_sai = sai[node];
  for (int j = lane; j < 256; j += 32) {
    float l = silu_f(my_sai + saj[bb * 256 + j]);
    if (j == i) l -= 1e10f;                        // diagonal mask (matches ref)
    S.att[j] = l;
  }
  __syncthreads();                                 // the only block barrier

  if (lane == 0) {                                 // deterministic softmax (in-wave LDS order)
    float m = -1e30f;
    for (int j = 0; j < 256; ++j) m = fmaxf(m, S.att[j]);
    float s = 0.f;
    for (int j = 0; j < 256; ++j) { float e = __expf(S.att[j] - m); S.att[j] = e; s += e; }
    const float inv = fast_rcp(s);
    for (int j = 0; j < 256; ++j) S.att[j] *= inv;
  }

  const int j16 = lane & 15, hi8 = lane >> 4;
  const float xi0 = x[node * 3 + 0], xi1 = x[node * 3 + 1], xi2 = x[node * 3 + 2];
  const float cb2 = cm_b2[0];
  const float m0   = 0.006737946999085467f;        // exp(-5)
  const float dmn  = (1.0f - m0) / 49.0f;
  const float cc   = 0.04f * (1.0f - m0);
  const float beta = 1.0f / (cc * cc);

  float hagg[4][8] = {};
  float comba[2][3] = {};
  float xacc[3] = {0.f, 0.f, 0.f};

  for (int jt = 0; jt < 16; ++jt) {
    // Opaque zero: makes the weight/bias LDS reads loop-variant so they are
    // NOT hoisted into long-lived registers (keeps VGPR count low).
    int zo = 0;
    asm volatile("" : "+v"(zo));

    const int jb = jt * 16;
    const int jg = bb * 256 + jb + j16;            // this lane's edge node j
    // prefetch next tile's rows (wraps to a valid tile at jt==15)
    const int jn = bb * 256 + ((jt + 1) & 15) * 16 + j16;
    __builtin_prefetch(pb + jn * 64, 0, 0);
    __builtin_prefetch(eb + jn * 64, 0, 0);
    __builtin_prefetch(x + jn * 3, 0, 0);

    // ---- geometry (per lane; duplicated across feature halves) ----
    const float d0 = xi0 - x[jg * 3 + 0];
    const float d1 = xi1 - x[jg * 3 + 1];
    const float d2 = xi2 - x[jg * 3 + 2];
    const float dn2 = d0 * d0 + d1 * d1 + d2 * d2 + 1e-5f;
    const float rin = __builtin_amdgcn_rsqf(dn2);
    const float dn  = dn2 * rin;
    const float u0 = d0 * rin, u1 = d1 * rin, u2 = d2 * rin;
    if (lane < 16) { S.dxu[j16][0] = u0; S.dxu[j16][1] = u1; S.dxu[j16][2] = u2; }
    const float E = __expf(-dn);

    // ---- build t1^T B operands directly (K = rbf idx, N = edge) ----
    const float* pbrow = pb + jg * 64;
    v16h b_lo, b_hi;
#pragma unroll
    for (int t = 0; t < 16; ++t) {
      const int kb = ((t >> 3) << 4) + 8 * hi8 + (t & 7);
#pragma unroll
      for (int ks = 0; ks < 2; ++ks) {
        const int k = 32 * ks + kb;
        const float e = E - (m0 + (float)k * dmn);
        float v = (S.pai[zo + k] + pbrow[k]) * __expf(-beta * e * e);  // 0 for k>=50 (padded)
        if (k == 63) v = 1.0f;                     // bias row (b1 folded into W1 pack)
        if (ks == 0) b_lo[t] = (half_t)v; else b_hi[t] = (half_t)v;
      }
    }

    // ---- GEMM1: t2^T = silu(W1^T @ t1^T) (bias folded); D -> B in-loop ----
    v16h s_lo, s_hi;
#pragma unroll
    for (int nt = 0; nt < 4; ++nt) {
      v8f c = {};
      c = wmma_f16(load_W(wpk, zo, 0, nt, 0, lane), b_lo, c);
      c = wmma_f16(load_W(wpk, zo, 0, nt, 1, lane), b_hi, c);
#pragma unroll
      for (int v = 0; v < 8; ++v) {
        const half_t hv = (half_t)silu_f(c[v]);
        if      (nt == 0) s_lo[v]     = hv;
        else if (nt == 1) s_lo[8 + v] = hv;
        else if (nt == 2) s_hi[v]     = hv;
        else              s_hi[8 + v] = hv;
      }
    }

    // ---- GEMM2: he^T = W2^T @ t2^T + b2; h_agg += att_j * he ----
    const float attv = S.att[jb + j16];
    v16h h_lo, h_hi;
#pragma unroll
    for (int nt = 0; nt < 4; ++nt) {
      v8f c = {};
      c = wmma_f16(load_W(wpk, zo, 1, nt, 0, lane), s_lo, c);
      c = wmma_f16(load_W(wpk, zo, 1, nt, 1, lane), s_hi, c);
      const int f0 = nt * 16 + 8 * hi8;
#pragma unroll
      for (int v = 0; v < 8; ++v) {
        const float hv = c[v] + b2l[zo + f0 + v];
        hagg[nt][v] += hv * attv;
        const half_t hh = (half_t)hv;
        if      (nt == 0) h_lo[v]     = hh;
        else if (nt == 1) h_lo[8 + v] = hh;
        else if (nt == 2) h_hi[v]     = hh;
        else              h_hi[8 + v] = hh;
      }
    }

    // ---- GEMM3: coeff^T = tanh(ew3^T @ he^T + ea_i + eb_j) -> LDS tile ----
    const float* ebrow = eb + jg * 64;
#pragma unroll
    for (int nt = 0; nt < 4; ++nt) {
      v8f c = {};
      c = wmma_f16(load_W(wpk, zo, 2, nt, 0, lane), h_lo, c);
      c = wmma_f16(load_W(wpk, zo, 2, nt, 1, lane), h_hi, c);
      const int f0 = nt * 16 + 8 * hi8;
#pragma unroll
      for (int v = 0; v < 8; ++v)
        S.stage[j16 * 64 + f0 + v] = tanh_f(c[v] + S.eai[zo + f0 + v] + ebrow[f0 + v]);
    }

    // ---- GEMM4: cw = silu(cm1^T @ he^T + cb1) . cm_w2 + cb2 ----
    float pr = 0.f;
#pragma unroll
    for (int nt = 0; nt < 4; ++nt) {
      v8f c = {};
      c = wmma_f16(load_W(wpk, zo, 3, nt, 0, lane), h_lo, c);
      c = wmma_f16(load_W(wpk, zo, 3, nt, 1, lane), h_hi, c);
      const int f0 = nt * 16 + 8 * hi8;
#pragma unroll
      for (int v = 0; v < 8; ++v)
        pr += silu_f(c[v] + cb1l[zo + f0 + v]) * w2l[zo + f0 + v];
    }
    const float cw = pr + __shfl_xor(pr, 16) + cb2;   // both halves -> full 64-feature dot
    if (hi8 == 0) { xacc[0] += d0 * cw; xacc[1] += d1 * cw; xacc[2] += d2 * cw; }

    // ---- comb[k][c] += coeff[k][j] * dxu[j][c]  (wave-internal LDS ordering) ----
    {
      const int f0 = lane, f1 = lane + 32;           // 2 features per lane
      float p00 = 0.f, p01 = 0.f, p02 = 0.f, p10 = 0.f, p11 = 0.f, p12 = 0.f;
#pragma unroll 4
      for (int e = 0; e < 16; ++e) {
        const float w0 = S.dxu[e][0], w1 = S.dxu[e][1], w2 = S.dxu[e][2];
        const float c0 = S.stage[e * 64 + f0];
        const float c1 = S.stage[e * 64 + f1];
        p00 += c0 * w0; p01 += c0 * w1; p02 += c0 * w2;
        p10 += c1 * w0; p11 += c1 * w1; p12 += c1 * w2;
      }
      comba[0][0] += p00; comba[0][1] += p01; comba[0][2] += p02;
      comba[1][0] += p10; comba[1][1] += p11; comba[1][2] += p12;
    }
  }

  // ---- writeback ----
#pragma unroll
  for (int nt = 0; nt < 4; ++nt)
#pragma unroll
    for (int v = 0; v < 8; ++v) {                    // sum over 16 edges (lanes in half)
      float s = hagg[nt][v];
      s += __shfl_xor(s, 1); s += __shfl_xor(s, 2);
      s += __shfl_xor(s, 4); s += __shfl_xor(s, 8);
      if (j16 == 0) hagg_o[node * 64 + nt * 16 + 8 * hi8 + v] = s;
    }
#pragma unroll
  for (int c = 0; c < 3; ++c) {
    comb_o[node * 192 + lane * 3 + c]        = comba[0][c];
    comb_o[node * 192 + (lane + 32) * 3 + c] = comba[1][c];
  }
#pragma unroll
  for (int c = 0; c < 3; ++c) {
    float s = xacc[c];
    s += __shfl_xor(s, 1); s += __shfl_xor(s, 2);
    s += __shfl_xor(s, 4); s += __shfl_xor(s, 8);
    if (lane == 0) xacc_o[node * 3 + c] = s;
  }
}

// ---------------------------------------------------------------------------
// K3: per-node finalize (pn/nm MLPs + coordinate update).
// ---------------------------------------------------------------------------
__global__ __launch_bounds__(64) void sake_node_post(
    const float* h, const float* x,
    const float* pn_w1, const float* pn_b1, const float* pn_w2, const float* pn_b2,
    const float* nm_w1, const float* nm_b1, const float* nm_w2, const float* nm_b2,
    const float* hagg, const float* comb, const float* xacc, float* out) {
  __shared__ float cn[64], u1[64], cat[192], u2[64];
  const int node = blockIdx.x;
  const int t = threadIdx.x;
  const float sc = 1.0f / 256.0f;

  const float q0 = comb[node * 192 + t * 3 + 0] * sc;
  const float q1 = comb[node * 192 + t * 3 + 1] * sc;
  const float q2 = comb[node * 192 + t * 3 + 2] * sc;
  cn[t] = q0 * q0 + q1 * q1 + q2 * q2;
  cat[t]      = h[node * 64 + t];
  cat[64 + t] = hagg[node * 64 + t];
  __syncthreads();

  float s = 0.f;
#pragma unroll 4
  for (int k = 0; k < 64; ++k) s += cn[k] * pn_w1[k * 64 + t];
  u1[t] = silu_f(s + pn_b1[t]);
  __syncthreads();

  s = 0.f;
#pragma unroll 4
  for (int k = 0; k < 64; ++k) s += u1[k] * pn_w2[k * 64 + t];
  cat[128 + t] = s + pn_b2[t];
  __syncthreads();

  s = 0.f;
#pragma unroll 4
  for (int m = 0; m < 192; ++m) s += cat[m] * nm_w1[m * 64 + t];
  u2[t] = silu_f(s + nm_b1[t]);
  __syncthreads();

  s = 0.f;
#pragma unroll 4
  for (int k = 0; k < 64; ++k) s += u2[k] * nm_w2[k * 64 + t];
  out[node * 64 + t] = s + nm_b2[t];

  if (t < 3) out[XOFF + node * 3 + t] = x[node * 3 + t] + xacc[node * 3 + t] * sc;
}

// ---------------------------------------------------------------------------
extern "C" void kernel_launch(void* const* d_in, const int* in_sizes, int n_in,
                              void* d_out, int out_size, void* d_ws, size_t ws_size,
                              hipStream_t stream) {
  (void)in_sizes; (void)n_in; (void)out_size; (void)ws_size;
  const float* h         = (const float*)d_in[0];
  const float* x         = (const float*)d_in[1];
  const float* df_in_w   = (const float*)d_in[2];
  const float* df_in_b   = (const float*)d_in[3];
  const float* df_out_w1 = (const float*)d_in[4];
  const float* df_out_b1 = (const float*)d_in[5];
  const float* df_out_w2 = (const float*)d_in[6];
  const float* df_out_b2 = (const float*)d_in[7];
  const float* ew_w      = (const float*)d_in[8];
  const float* ew_b      = (const float*)d_in[9];
  const float* pn_w1     = (const float*)d_in[10];
  const float* pn_b1     = (const float*)d_in[11];
  const float* pn_w2     = (const float*)d_in[12];
  const float* pn_b2     = (const float*)d_in[13];
  const float* nm_w1     = (const float*)d_in[14];
  const float* nm_b1     = (const float*)d_in[15];
  const float* nm_w2     = (const float*)d_in[16];
  const float* nm_b2     = (const float*)d_in[17];
  const float* cm_w1     = (const float*)d_in[18];
  const float* cm_b1     = (const float*)d_in[19];
  const float* cm_w2     = (const float*)d_in[20];
  const float* cm_b2     = (const float*)d_in[21];
  const float* sa_w      = (const float*)d_in[22];

  float* f = (float*)d_ws;
  float* pa   = f; f += TOTN * 64;
  float* pb   = f; f += TOTN * 64;
  float* ea   = f; f += TOTN * 64;
  float* eb   = f; f += TOTN * 64;
  float* sai  = f; f += TOTN;
  float* saj  = f; f += TOTN;
  half_t* wp  = (half_t*)f; f += (4 * 4096) / 2;   // 16384 halves
  float* hagg = f; f += TOTN * 64;
  float* comb = f; f += TOTN * 192;
  float* xacc = f; f += TOTN * 3;

  sake_pack_weights<<<1, 256, 0, stream>>>(df_out_w1, df_out_b1, df_out_w2, ew_w, cm_w1, wp);
  sake_node_pre<<<TOTN, 64, 0, stream>>>(h, df_in_w, df_in_b, ew_w, ew_b, sa_w,
                                         pa, pb, ea, eb, sai, saj);
  sake_edges<<<TOTN / 4, 128, 0, stream>>>(x, df_out_b2, cm_b1, cm_w2, cm_b2,
                                           pa, pb, ea, eb, sai, saj, wp,
                                           hagg, comb, xacc);
  sake_node_post<<<TOTN, 64, 0, stream>>>(h, x, pn_w1, pn_b1, pn_w2, pn_b2,
                                          nm_w1, nm_b1, nm_w2, nm_b2,
                                          hagg, comb, xacc, (float*)d_out);
}
